// SabiaNetwork_79207786872899
// MI455X (gfx1250) — matrix operand order repaired and checked by reading, not compiled, not run on hardware
//
#include <hip/hip_runtime.h>
#include <cstdint>

// ---------------------------------------------------------------------------
// SabiaNetwork forward, CDNA5 (gfx1250, wave32) WMMA implementation.
//
// Pipeline:
//  1) prep_kernel   : pack W1 (10x100, K-pad 32, N-pad 128), W2*0.1 (100x128,
//                     K-pad 128) and W_in (64x64) into wave32 WMMA-B fragments.
//  2) xw_kernel     : xt = x @ W_in           via v_wmma_f32_16x16x32_f16
//  3) xs_kernel     : Xs[g] = sum_{batch=g} x  (LDS accumulate + global flush)
//  4) edge_kernel   : stage W1/W2 fragments in LDS once per block; then per
//                     16-edge tile per wave:
//                       geometry -> rb -> WMMA (rb@W1) -> silu -> LDS
//                       transpose -> WMMA (h@W2) -> messages -> per-graph
//                       LDS accumulation (pooling folded into aggregation)
//  5) final_kernel  : pooled = [G0/sqrt(16) @ Wouts + Xs @ Wsc,
//                               G1/sqrt(16) . Woutv] / sqrt(1000)
// ---------------------------------------------------------------------------

typedef __attribute__((ext_vector_type(16))) _Float16 v16h;
typedef __attribute__((ext_vector_type(8)))  _Float16 v8h;
typedef __attribute__((ext_vector_type(8)))  float    v8f;
typedef __attribute__((ext_vector_type(4)))  float    v4f;

#define N_NODES 50000
#define N_EDGES 800000
#define N_GRAPHS 50
#define GSTR 256            // per-graph accumulator stride: 64 (G0) + 192 (G1)

__device__ __forceinline__ v8f wmma16(v16h a, v16h b, v8f c) {
  // D = A(16x32,f16) x B(32x16,f16) + C(16x16,f32)
  return __builtin_amdgcn_wmma_f32_16x16x32_f16(
      false, a, false, b, (short)0, c, false, false);
}

union H2U { _Float16 h; unsigned short u; };

// ---------------------------------------------------------------------------
// 1) Fragment packing.  B-fragment layout (16-bit, 32x16): lane -> column
//    N = lane&15; half=lane>>4; element j (0..15) -> K = 16*(j>>3) + 8*half + (j&7).
//    Stored as 8 dwords per lane, lane-contiguous for coalesced v16h loads.
// ---------------------------------------------------------------------------
__global__ __launch_bounds__(256) void prep_kernel(
    const float* __restrict__ W1, const float* __restrict__ W2,
    const float* __restrict__ Win,
    uint32_t* __restrict__ w1f, uint32_t* __restrict__ w2f,
    uint32_t* __restrict__ winf)
{
  int tid = blockIdx.x * 256 + threadIdx.x;
  if (tid >= 8192) return;
  const int d    = tid & 7;
  const int lane = (tid >> 3) & 31;
  const int half = lane >> 4;
  const int n16  = lane & 15;

  // W2 fragments: [kt(4)][nt(8)][lane][8 dwords], pre-scaled by 1/sqrt(H)=0.1
  {
    const int nt = (tid >> 8) & 7;
    const int kt = tid >> 11;
    const int n  = nt * 16 + n16;
    uint32_t pack = 0;
    #pragma unroll
    for (int s = 0; s < 2; ++s) {
      const int j = 2 * d + s;
      const int k = kt * 32 + ((j >> 3) << 4) + half * 8 + (j & 7);
      H2U c; c.h = (_Float16)((k < 100) ? W2[k * 128 + n] * 0.1f : 0.0f);
      pack |= ((uint32_t)c.u) << (16 * s);
    }
    w2f[tid] = pack;
  }
  if (tid < 2048) {
    // W1 fragments: [nt(8)][lane][8]; sqrt(B) and 1/sqrt(B) cancel -> raw W1.
    {
      const int nt = (tid >> 8) & 7;
      const int n  = nt * 16 + n16;
      uint32_t pack = 0;
      #pragma unroll
      for (int s = 0; s < 2; ++s) {
        const int j = 2 * d + s;
        const int k = ((j >> 3) << 4) + half * 8 + (j & 7);
        H2U c; c.h = (_Float16)((k < 10 && n < 100) ? W1[k * 100 + n] : 0.0f);
        pack |= ((uint32_t)c.u) << (16 * s);
      }
      w1f[tid] = pack;
    }
    // W_in fragments: [kt(2)][nt(4)][lane][8]
    {
      const int nt = (tid >> 8) & 3;
      const int kt = (tid >> 10) & 1;
      const int n  = nt * 16 + n16;
      uint32_t pack = 0;
      #pragma unroll
      for (int s = 0; s < 2; ++s) {
        const int j = 2 * d + s;
        const int k = kt * 32 + ((j >> 3) << 4) + half * 8 + (j & 7);
        H2U c; c.h = (_Float16)Win[k * 64 + n];
        pack |= ((uint32_t)c.u) << (16 * s);
      }
      winf[tid] = pack;
    }
  }
}

// ---------------------------------------------------------------------------
// 2) xt = x @ W_in  (50000x64 @ 64x64).  One 16-row tile per wave.
//    A-fragment layout (16-bit, 16x32): lane -> row M = lane&15;
//    element j -> K = 16*(j>>3) + 8*half + (j&7)  (contiguous runs of 8).
// ---------------------------------------------------------------------------
__global__ __launch_bounds__(256) void xw_kernel(
    const float* __restrict__ x, const uint32_t* __restrict__ winf,
    float* __restrict__ xt)
{
  const int lane = threadIdx.x & 31;
  const int w    = threadIdx.x >> 5;
  const int half = lane >> 4;
  const int m    = lane & 15;
  const int tile = blockIdx.x * 8 + w;          // wave-uniform
  if (tile >= N_NODES / 16) return;
  const int row = tile * 16 + m;

  v8f c[4];
  #pragma unroll
  for (int nt = 0; nt < 4; ++nt) c[nt] = (v8f)0.0f;

  #pragma unroll
  for (int kt = 0; kt < 2; ++kt) {
    const float* xr = x + (size_t)row * 64 + kt * 32;
    v4f a0 = *(const v4f*)(xr + half * 8);
    v4f a1 = *(const v4f*)(xr + half * 8 + 4);
    v4f a2 = *(const v4f*)(xr + 16 + half * 8);
    v4f a3 = *(const v4f*)(xr + 16 + half * 8 + 4);
    v16h A;
    #pragma unroll
    for (int j = 0; j < 4; ++j) {
      A[j]      = (_Float16)a0[j];
      A[4 + j]  = (_Float16)a1[j];
      A[8 + j]  = (_Float16)a2[j];
      A[12 + j] = (_Float16)a3[j];
    }
    #pragma unroll
    for (int nt = 0; nt < 4; ++nt) {
      v16h B = *(const v16h*)(winf + ((kt * 4 + nt) * 32 + lane) * 8);
      c[nt] = wmma16(A, B, c[nt]);
    }
  }
  // D layout: value (M = v + 8*half, N = nt*16 + m)
  #pragma unroll
  for (int nt = 0; nt < 4; ++nt)
    #pragma unroll
    for (int v = 0; v < 8; ++v)
      xt[(size_t)(tile * 16 + v + 8 * half) * 64 + nt * 16 + m] = c[nt][v];
}

// ---------------------------------------------------------------------------
// 3) Xs[g][f] = sum over nodes in graph g of x[n][f]
// ---------------------------------------------------------------------------
__global__ __launch_bounds__(256) void xs_kernel(
    const float* __restrict__ x, const int* __restrict__ batch,
    float* __restrict__ XsG)
{
  __shared__ float acc[N_GRAPHS * 64];
  const int tid = threadIdx.x;
  for (int i = tid; i < N_GRAPHS * 64; i += 256) acc[i] = 0.0f;
  __syncthreads();

  const int total = N_NODES * 64;
  for (int idx = blockIdx.x * 256 + tid; idx < total;
       idx += gridDim.x * 256) {
    const int nid = idx >> 6;
    const int f   = idx & 63;
    atomicAdd(&acc[batch[nid] * 64 + f], x[idx]);
  }
  __syncthreads();
  for (int i = tid; i < N_GRAPHS * 64; i += 256)
    unsafeAtomicAdd(&XsG[i], acc[i]);
}

// ---------------------------------------------------------------------------
// 4) Edge kernel: the heavy one.  W1/W2 fragments staged in LDS per block.
// ---------------------------------------------------------------------------
__global__ __launch_bounds__(256) void edge_kernel(
    const float* __restrict__ pos, const int* __restrict__ ei,
    const float* __restrict__ shift, const float* __restrict__ lat,
    const int* __restrict__ batch, const float* __restrict__ xt,
    const uint32_t* __restrict__ w1f, const uint32_t* __restrict__ w2f,
    float* __restrict__ Gg)
{
  __shared__ float    Gacc[N_GRAPHS * GSTR];               // 51200 B
  __shared__ alignas(32) uint32_t w1lds[2048];             //  8192 B
  __shared__ alignas(32) uint32_t w2lds[8192];             // 32768 B
  __shared__ alignas(16) _Float16 hbuf[8][16 * 32];        //  8192 B
  __shared__ int   esrc[8][16];
  __shared__ int   eg[8][16];
  __shared__ float esh[8][16][3];

  const int tid = threadIdx.x;
  for (int i = tid; i < N_GRAPHS * GSTR; i += 256) Gacc[i] = 0.0f;
  for (int i = tid; i < 2048; i += 256) w1lds[i] = w1f[i];
  for (int i = tid; i < 8192; i += 256) w2lds[i] = w2f[i];
  __syncthreads();

  const int lane = tid & 31;
  const int w    = tid >> 5;
  const int half = lane >> 4;
  const int m    = lane & 15;
  const int waveGlobal = blockIdx.x * 8 + w;
  const int nWaves     = gridDim.x * 8;
  const int NUNITS     = N_EDGES / 16;
  const float SQ3 = 1.7320508075688772f;

  _Float16* hb = &hbuf[w][0];

  for (int u = waveGlobal; u < NUNITS; u += nWaves) {
    // -------- geometry (each half computes edge m redundantly) --------
    const int e = u * 16 + m;
    __builtin_prefetch(ei + e + 16 * nWaves, 0, 1);       // next unit's edges
    const int s  = ei[e];
    const int dd = ei[N_EDGES + e];
    const int gl = batch[s];
    const float shx = shift[e * 3 + 0];
    const float shy = shift[e * 3 + 1];
    const float shz = shift[e * 3 + 2];
    const float* Lm = lat + gl * 9;
    const float vx = pos[dd * 3 + 0] - pos[s * 3 + 0]
                   + shx * Lm[0] + shy * Lm[3] + shz * Lm[6];
    const float vy = pos[dd * 3 + 1] - pos[s * 3 + 1]
                   + shx * Lm[1] + shy * Lm[4] + shz * Lm[7];
    const float vz = pos[dd * 3 + 2] - pos[s * 3 + 2]
                   + shx * Lm[2] + shy * Lm[5] + shz * Lm[8];
    const float rr  = sqrtf(vx * vx + vy * vy + vz * vz);
    const float inv = 1.0f / rr;
    if (half == 0) {
      esrc[w][m] = s;
      eg[w][m]   = batch[dd];
      esh[w][m][0] = SQ3 * vy * inv;   // sh1 = sqrt(3) * n[(1,2,0)]
      esh[w][m][1] = SQ3 * vz * inv;
      esh[w][m][2] = SQ3 * vx * inv;
    }
    // radial basis (sqrt(B) scaling cancels with /sqrt(B) in rb@W1)
    float rb[10];
    #pragma unroll
    for (int k = 0; k < 10; ++k) {
      const float t = rr * (11.0f / 5.0f) - (float)(k + 1);
      const float cc = __cosf(1.5707963267948966f * t);
      rb[k] = (t > -1.0f && t < 1.0f) ? cc : 0.0f;
    }
    // A1 fragment (K = 10, zero-padded to 32)
    v16h a1;
    #pragma unroll
    for (int j = 0; j < 16; ++j) {
      const int K = ((j >> 3) << 4) + (half << 3) + (j & 7);
      a1[j] = (K < 10) ? (_Float16)rb[K] : (_Float16)0.0f;
    }

    v8f c2[8];
    #pragma unroll
    for (int i = 0; i < 8; ++i) c2[i] = (v8f)0.0f;

    // -------- fused MLP: (rb@W1) -> silu -> LDS transpose -> (h@W2) ------
    #pragma unroll
    for (int kt = 0; kt < 4; ++kt) {
      #pragma unroll
      for (int i = 0; i < 2; ++i) {
        const int t = kt * 2 + i;
        v16h b1 = *(const v16h*)(w1lds + (t * 32 + lane) * 8);
        v8f c1 = wmma16(a1, b1, (v8f)0.0f);
        #pragma unroll
        for (int v = 0; v < 8; ++v) {
          const float hv = c1[v];
          const float sv = hv / (1.0f + __expf(-hv));       // silu
          hb[(v + 8 * half) * 32 + i * 16 + m] = (_Float16)sv;
        }
      }
      __builtin_amdgcn_wave_barrier();
      // A2 fragment for this k-tile from LDS (M<->K transpose)
      v8h lo = *(const v8h*)(hb + m * 32 + half * 8);
      v8h hi = *(const v8h*)(hb + m * 32 + 16 + half * 8);
      v16h a2 = __builtin_shufflevector(lo, hi, 0, 1, 2, 3, 4, 5, 6, 7,
                                        8, 9, 10, 11, 12, 13, 14, 15);
      #pragma unroll
      for (int nt = 0; nt < 8; ++nt) {
        v16h b2 = *(const v16h*)(w2lds + ((kt * 8 + nt) * 32 + lane) * 8);
        c2[nt] = wmma16(a2, b2, c2[nt]);
      }
      __builtin_amdgcn_wave_barrier();
    }

    // -------- messages + per-graph pooling (LDS atomics) -----------------
    // w0 = c2[0..3] (f = nt*16+m), w1 = c2[4..7] (same f) share xs gather.
    #pragma unroll
    for (int nt = 0; nt < 4; ++nt) {
      const int f = nt * 16 + m;
      #pragma unroll
      for (int v = 0; v < 8; ++v) {
        const int M = v + 8 * half;
        const int ss = esrc[w][M];
        const int gg = eg[w][M];
        const float xs = xt[(size_t)ss * 64 + f];
        atomicAdd(&Gacc[gg * GSTR + f], c2[nt][v] * xs);      // msg0
        const float wv = c2[nt + 4][v] * xs;                  // msg1 radial
        float* gp = &Gacc[gg * GSTR + 64 + f * 3];
        atomicAdd(gp + 0, wv * esh[w][M][0]);
        atomicAdd(gp + 1, wv * esh[w][M][1]);
        atomicAdd(gp + 2, wv * esh[w][M][2]);
      }
    }
  }

  __syncthreads();
  for (int i = tid; i < N_GRAPHS * GSTR; i += 256)
    unsafeAtomicAdd(&Gg[i], Gacc[i]);
}

// ---------------------------------------------------------------------------
// 5) Tiny output GEMMs on pooled (50x256) accumulators.
// ---------------------------------------------------------------------------
__global__ __launch_bounds__(256) void final_kernel(
    const float* __restrict__ Gg, const float* __restrict__ XsG,
    const float* __restrict__ Wouts, const float* __restrict__ Wsc,
    const float* __restrict__ Woutv, float* __restrict__ out)
{
  const int tid = blockIdx.x * 256 + threadIdx.x;
  if (tid >= N_GRAPHS * 56) return;
  const int g = tid / 56;
  const int j = tid % 56;
  const float innorm = 0.031622776601683794f;   // 1/sqrt(1000)
  const float invnn  = 0.25f;                   // 1/sqrt(16)
  float acc = 0.0f;
  if (j < 32) {
    #pragma unroll 8
    for (int f = 0; f < 64; ++f)
      acc += Gg[g * GSTR + f] * invnn * Wouts[f * 32 + j]
           + XsG[g * 64 + f] * Wsc[f * 32 + j];
  } else {
    const int o = (j - 32) / 3;
    const int c = (j - 32) % 3;
    #pragma unroll 8
    for (int f = 0; f < 64; ++f)
      acc += Gg[g * GSTR + 64 + f * 3 + c] * Woutv[f * 8 + o];
    acc *= invnn;
  }
  out[tid] = acc * innorm;
}

// ---------------------------------------------------------------------------
extern "C" void kernel_launch(void* const* d_in, const int* in_sizes, int n_in,
                              void* d_out, int out_size, void* d_ws,
                              size_t ws_size, hipStream_t stream)
{
  const float* pos   = (const float*)d_in[0];
  const float* x     = (const float*)d_in[1];
  const int*   ei    = (const int*)  d_in[2];
  const float* shift = (const float*)d_in[3];
  const float* lat   = (const float*)d_in[4];
  const int*   batch = (const int*)  d_in[5];
  const float* W1    = (const float*)d_in[6];
  const float* W2    = (const float*)d_in[7];
  const float* Win   = (const float*)d_in[8];
  const float* Wsc   = (const float*)d_in[9];
  const float* Wouts = (const float*)d_in[10];
  const float* Woutv = (const float*)d_in[11];
  float* out = (float*)d_out;

  // Workspace layout (all offsets 32B-aligned)
  float*    xt   = (float*)d_ws;                 // 3,200,000 f32 (12.8 MB)
  float*    Gg   = xt + 3200000;                 // 12,800 f32
  float*    XsG  = Gg + N_GRAPHS * GSTR;         // 3,200 f32
  uint32_t* w1f  = (uint32_t*)(XsG + 3200);      // 2,048 dwords
  uint32_t* w2f  = w1f + 2048;                   // 8,192 dwords
  uint32_t* winf = w2f + 8192;                   // 2,048 dwords

  // Zero the pooled accumulators (Gg and XsG are contiguous).
  hipMemsetAsync(Gg, 0, (size_t)(N_GRAPHS * GSTR + N_GRAPHS * 64) * 4, stream);

  prep_kernel <<<32,  256, 0, stream>>>(W1, W2, Win, w1f, w2f, winf);
  xw_kernel   <<<391, 256, 0, stream>>>(x, winf, xt);        // 3125 tiles
  xs_kernel   <<<128, 256, 0, stream>>>(x, batch, XsG);
  edge_kernel <<<512, 256, 0, stream>>>(pos, ei, shift, lat, batch, xt,
                                        w1f, w2f, Gg);
  final_kernel<<<11,  256, 0, stream>>>(Gg, XsG, Wouts, Wsc, Woutv, out);
}
